// TransformerDecoder_85005992722939
// MI455X (gfx1250) — compile-verified, matrix operand order checked by reading, not run on hardware
//
#include <hip/hip_runtime.h>
#include <hip/hip_bf16.h>
#include <math.h>

// ---------------- problem constants ----------------
constexpr int cB = 32, cS = 256, cD = 512, cH = 8, cDK = 64, cF = 2048,
              cL = 6, cFEAT = 9;
constexpr int cM   = cB * cS;   // 8192 tokens
constexpr int cHDK = cH * cDK;  // 512

typedef __attribute__((ext_vector_type(16))) _Float16 v16h;
typedef __attribute__((ext_vector_type(8)))  _Float16 v8h;
typedef __attribute__((ext_vector_type(8)))  float    v8f;

// ===================================================================
// Generic batched WMMA GEMM:  C[M,N] = alpha * A[M,K] @ Bt[N,K]^T + bias
//   A, Bt fp16; C fp32 (optional), Ch fp16 (optional); act=1 -> relu.
//   Batch z: outer = z/innerB, inner = z%innerB with separate strides.
//   Block = 128 threads (4 waves). Wave w owns rows
//   [by*128 + w*32, +32) as 2 M-tiles and cols [bx*64, +64) as 4 N-tiles.
//   All 6 fragment loads (2 A + 4 B, 12x b128) are issued before the
//   8-WMMA group so the scheduler can overlap loads with matrix ops on
//   partial loadcnt waits instead of full drains.
// ===================================================================
__global__ __launch_bounds__(128) void k_gemm_wmma(
    const _Float16* __restrict__ A, const _Float16* __restrict__ Bt,
    float* __restrict__ C, _Float16* __restrict__ Ch,
    const float* __restrict__ bias,
    int Kk, int lda, int ldb, int ldc,
    long sAo, long sAi, long sBo, long sBi, long sCo, long sCi, int innerB,
    float alpha, int act)
{
    const int z  = blockIdx.z;
    const int bo = z / innerB, bi = z % innerB;
    const _Float16* Ab = A  + (size_t)bo * sAo + (size_t)bi * sAi;
    const _Float16* Bb = Bt + (size_t)bo * sBo + (size_t)bi * sBi;
    const long cbase = (long)bo * sCo + (long)bi * sCi;

    const int lane = threadIdx.x & 31;
    const int wave = threadIdx.x >> 5;
    const int m0 = blockIdx.y * 128 + wave * 32;   // 2 M-tiles: m0, m0+16
    const int n0 = blockIdx.x * 64;                // 4 N-tiles
    const int row = lane & 15;
    const int kg  = lane >> 4;   // K-group: lanes 16..31 hold K+8 / K+24

    v8f acc[2][4];
#pragma unroll
    for (int mi = 0; mi < 2; ++mi)
#pragma unroll
        for (int t = 0; t < 4; ++t) acc[mi][t] = (v8f){};

    const _Float16* arow0 = Ab + (size_t)(m0 + row) * lda + kg * 8;
    const _Float16* arow1 = arow0 + (size_t)16 * lda;
    const _Float16* brow0 = Bb + (size_t)(n0 + row) * ldb + kg * 8;

    for (int k = 0; k < Kk; k += 32) {
        // ---- issue ALL fragment loads first (one big clause) ----
        v8h alo0 = *(const v8h*)(arow0 + k);
        v8h ahi0 = *(const v8h*)(arow0 + k + 16);
        v8h alo1 = *(const v8h*)(arow1 + k);
        v8h ahi1 = *(const v8h*)(arow1 + k + 16);
        v8h blo[4], bhi[4];
#pragma unroll
        for (int t = 0; t < 4; ++t) {
            const _Float16* br = brow0 + (size_t)(t * 16) * ldb + k;
            blo[t] = *(const v8h*)(br);
            bhi[t] = *(const v8h*)(br + 16);
        }
        if (k + 32 < Kk) {
            __builtin_prefetch((const void*)(arow0 + k + 32), 0, 1);
            __builtin_prefetch((const void*)(arow1 + k + 32), 0, 1);
        }

        v16h af[2];
#pragma unroll
        for (int i = 0; i < 8; ++i) {
            af[0][i] = alo0[i]; af[0][8 + i] = ahi0[i];
            af[1][i] = alo1[i]; af[1][8 + i] = ahi1[i];
        }

#pragma unroll
        for (int t = 0; t < 4; ++t) {
            v16h bf;
#pragma unroll
            for (int i = 0; i < 8; ++i) { bf[i] = blo[t][i]; bf[8 + i] = bhi[t][i]; }
            acc[0][t] = __builtin_amdgcn_wmma_f32_16x16x32_f16(
                false, af[0], false, bf, (short)0, acc[0][t], false, false);
            acc[1][t] = __builtin_amdgcn_wmma_f32_16x16x32_f16(
                false, af[1], false, bf, (short)0, acc[1][t], false, false);
        }
    }

    const int nloc = lane & 15;
    const int mo   = (lane >> 4) * 8;
#pragma unroll
    for (int mi = 0; mi < 2; ++mi) {
#pragma unroll
        for (int t = 0; t < 4; ++t) {
            const int ncol = n0 + t * 16 + nloc;
            const float bv = bias ? bias[ncol] : 0.f;
#pragma unroll
            for (int r = 0; r < 8; ++r) {
                long idx = cbase + (long)(m0 + mi * 16 + mo + r) * ldc + ncol;
                float v = acc[mi][t][r] * alpha + bv;
                if (act) v = fmaxf(v, 0.f);
                if (C)  C[idx]  = v;
                if (Ch) Ch[idx] = (_Float16)v;
            }
        }
    }
}

// ===================================================================
// Weight prep kernels (fp32 -> fp16, transposed to [N][K] for WMMA-B)
// ===================================================================
// wq/wk/wv: [L,H,D,DK] -> per layer [H*DK][D]
__global__ void k_prep_hdk(const float* __restrict__ in,
                           _Float16* __restrict__ out, int total)
{
    int gid = blockIdx.x * blockDim.x + threadIdx.x;
    if (gid >= total) return;
    int kk = gid % cDK; int t = gid / cDK;
    int d  = t % cD;    t /= cD;
    int h  = t % cH;    int l = t / cH;
    out[(((size_t)l * cHDK) + h * cDK + kk) * cD + d] = (_Float16)in[gid];
}

// generic [L,K,N] -> per layer [N][K]
__global__ void k_prep_kn(const float* __restrict__ in,
                          _Float16* __restrict__ out,
                          int Kd, int Nd, int total)
{
    int gid = blockIdx.x * blockDim.x + threadIdx.x;
    if (gid >= total) return;
    int n = gid % Nd; int t = gid / Nd;
    int k = t % Kd;   int l = t / Kd;
    out[((size_t)l * Nd + n) * Kd + k] = (_Float16)in[gid];
}

// plain fp32 -> fp16
__global__ void k_cvt(const float* __restrict__ in,
                      _Float16* __restrict__ out, int n)
{
    int gid = blockIdx.x * blockDim.x + threadIdx.x;
    if (gid < n) out[gid] = (_Float16)in[gid];
}

// V [B,S,H*DK] fp16 -> VT [B,H,DK,S] fp16
__global__ void k_vt(const _Float16* __restrict__ V,
                     _Float16* __restrict__ VT)
{
    int gid = blockIdx.x * blockDim.x + threadIdx.x;
    if (gid >= cM * cHDK) return;
    int dk = gid % cDK; int t = gid / cDK;
    int h  = t % cH;    t /= cH;
    int s  = t % cS;    int b = t / cS;
    VT[(((size_t)b * cH + h) * cDK + dk) * cS + s] = V[gid];
}

// ===================================================================
// Embedding: x = (tgt*mask)@w_f2h + b + pos_emb + emb_time/gender/race
// ===================================================================
__global__ void k_embed(const float* __restrict__ tgt,
                        const float* __restrict__ mask,
                        const float* __restrict__ w,
                        const float* __restrict__ bias,
                        const int* __restrict__ timei,
                        const int* __restrict__ gender,
                        const int* __restrict__ race,
                        const float* __restrict__ et,
                        const float* __restrict__ eg,
                        const float* __restrict__ er,
                        float* __restrict__ X, _Float16* __restrict__ XH)
{
    int gid = blockIdx.x * blockDim.x + threadIdx.x;
    if (gid >= cM * cD) return;
    int d   = gid % cD;
    int row = gid / cD;
    int b   = row / cS;
    int s   = row % cS;

    float acc = bias[d];
    const float* tg = tgt  + (size_t)row * cFEAT;
    const float* mk = mask + (size_t)row * cFEAT;
#pragma unroll
    for (int f = 0; f < cFEAT; ++f) acc += tg[f] * mk[f] * w[f * cD + d];

    float phase = (float)s / powf(1e4f, (float)d / (float)cD);
    acc += (d & 1) ? cosf(phase) : sinf(phase);
    acc += et[timei[row] * cD + d];
    acc += eg[gender[b] * cD + d];
    acc += er[race[b]  * cD + d];

    X[gid]  = acc;
    XH[gid] = (_Float16)acc;
}

// ===================================================================
// Row softmax over S=256, one wave per row, writes fp16 probabilities
// ===================================================================
__global__ __launch_bounds__(32) void k_softmax(const float* __restrict__ Sc,
                                                _Float16* __restrict__ P)
{
    const int row  = blockIdx.x;               // B*H*S rows
    const int lane = threadIdx.x;
    const float* r = Sc + (size_t)row * cS;
    _Float16*    p = P  + (size_t)row * cS;

    float v[8];
    float m = -1e30f;
#pragma unroll
    for (int i = 0; i < 8; ++i) { v[i] = r[lane + i * 32]; m = fmaxf(m, v[i]); }
#pragma unroll
    for (int off = 16; off > 0; off >>= 1) m = fmaxf(m, __shfl_xor(m, off, 32));

    float s = 0.f;
#pragma unroll
    for (int i = 0; i < 8; ++i) { v[i] = __expf(v[i] - m); s += v[i]; }
#pragma unroll
    for (int off = 16; off > 0; off >>= 1) s += __shfl_xor(s, off, 32);

    const float inv = 1.f / s;
#pragma unroll
    for (int i = 0; i < 8; ++i) p[lane + i * 32] = (_Float16)(v[i] * inv);
}

// ===================================================================
// Fused residual + LayerNorm over D=512, one wave per row;
// writes fp32 back to X and fp16 copy to XH.
// ===================================================================
__global__ __launch_bounds__(32) void k_add_ln(float* __restrict__ X,
                                               const float* __restrict__ Y,
                                               const float* __restrict__ g,
                                               const float* __restrict__ b,
                                               _Float16* __restrict__ XH)
{
    const int row  = blockIdx.x;
    const int lane = threadIdx.x;
    const size_t base = (size_t)row * cD;

    float v[16];
    float s = 0.f;
#pragma unroll
    for (int i = 0; i < 16; ++i) {
        int idx = lane + i * 32;
        v[i] = X[base + idx] + Y[base + idx];
        s += v[i];
    }
#pragma unroll
    for (int off = 16; off > 0; off >>= 1) s += __shfl_xor(s, off, 32);
    const float mean = s * (1.f / cD);

    float s2 = 0.f;
#pragma unroll
    for (int i = 0; i < 16; ++i) { float d = v[i] - mean; s2 += d * d; }
#pragma unroll
    for (int off = 16; off > 0; off >>= 1) s2 += __shfl_xor(s2, off, 32);
    const float rs = rsqrtf(s2 * (1.f / cD) + 1e-5f);

#pragma unroll
    for (int i = 0; i < 16; ++i) {
        int idx = lane + i * 32;
        float o = (v[i] - mean) * rs * g[idx] + b[idx];
        X[base + idx]  = o;
        XH[base + idx] = (_Float16)o;
    }
}

// ===================================================================
// Output heads: sigmoid(X @ w_out + b_out), sigmoid(X @ w_m + b_m)
// ===================================================================
__global__ void k_head(const float* __restrict__ X,
                       const float* __restrict__ w_out,
                       const float* __restrict__ b_out,
                       const float* __restrict__ w_m,
                       const float* __restrict__ b_m,
                       float* __restrict__ out, float* __restrict__ mout)
{
    int gid = blockIdx.x * blockDim.x + threadIdx.x;
    if (gid >= cM * cFEAT * 2) return;
    int which = gid / (cM * cFEAT);
    int r     = gid % (cM * cFEAT);
    int row   = r / cFEAT;
    int f     = r % cFEAT;

    const float* w  = which ? w_m : w_out;
    const float* bb = which ? b_m : b_out;
    const float* xr = X + (size_t)row * cD;
    float acc = bb[f];
    for (int k = 0; k < cD; ++k) acc += xr[k] * w[k * cFEAT + f];
    float sg = 1.f / (1.f + __expf(-acc));
    (which ? mout : out)[r] = sg;
}

// ===================================================================
// Host orchestration
// ===================================================================
extern "C" void kernel_launch(void* const* d_in, const int* in_sizes, int n_in,
                              void* d_out, int out_size, void* d_ws, size_t ws_size,
                              hipStream_t stream)
{
    (void)in_sizes; (void)n_in; (void)out_size; (void)ws_size;

    const float* tgt      = (const float*)d_in[0];
    const float* memory   = (const float*)d_in[1];
    const int*   timei    = (const int*)  d_in[2];
    const int*   gender   = (const int*)  d_in[3];
    const int*   race     = (const int*)  d_in[4];
    const float* maskp    = (const float*)d_in[5];
    const float* w_f2h    = (const float*)d_in[6];
    const float* b_f2h    = (const float*)d_in[7];
    const float* emb_time = (const float*)d_in[8];
    const float* emb_gen  = (const float*)d_in[9];
    const float* emb_race = (const float*)d_in[10];
    const float* wq1 = (const float*)d_in[11]; const float* bq1 = (const float*)d_in[12];
    const float* wk1 = (const float*)d_in[13]; const float* bk1 = (const float*)d_in[14];
    const float* wv1 = (const float*)d_in[15]; const float* bv1 = (const float*)d_in[16];
    const float* wo1 = (const float*)d_in[17]; const float* bo1 = (const float*)d_in[18];
    const float* g1  = (const float*)d_in[19]; const float* be1 = (const float*)d_in[20];
    const float* wq2 = (const float*)d_in[21]; const float* bq2 = (const float*)d_in[22];
    const float* wk2 = (const float*)d_in[23]; const float* bk2 = (const float*)d_in[24];
    const float* wv2 = (const float*)d_in[25]; const float* bv2 = (const float*)d_in[26];
    const float* wo2 = (const float*)d_in[27]; const float* bo2 = (const float*)d_in[28];
    const float* g2  = (const float*)d_in[29]; const float* be2 = (const float*)d_in[30];
    const float* w1f = (const float*)d_in[31]; const float* b1f = (const float*)d_in[32];
    const float* w2f = (const float*)d_in[33]; const float* b2f = (const float*)d_in[34];
    const float* gf  = (const float*)d_in[35]; const float* bff = (const float*)d_in[36];
    const float* w_out = (const float*)d_in[37]; const float* b_out = (const float*)d_in[38];
    const float* w_m   = (const float*)d_in[39]; const float* b_m   = (const float*)d_in[40];

    // ---- workspace bump allocator ----
    size_t off = 0;
    auto alloc = [&](size_t bytes) -> void* {
        off = (off + 255) & ~(size_t)255;
        void* p = (char*)d_ws + off;
        off += bytes;
        return p;
    };

    float*    X   = (float*)   alloc((size_t)cM * cD * 4);        // activations fp32
    float*    Y   = (float*)   alloc((size_t)cM * cD * 4);        // branch output fp32
    float*    SC  = (float*)   alloc((size_t)cB * cH * cS * cS * 4); // scores fp32
    _Float16* XH  = (_Float16*)alloc((size_t)cM * cD * 2);
    _Float16* MH  = (_Float16*)alloc((size_t)cM * cD * 2);        // memory fp16
    _Float16* QH  = (_Float16*)alloc((size_t)cM * cHDK * 2);
    _Float16* KH  = (_Float16*)alloc((size_t)cM * cHDK * 2);
    _Float16* VH  = (_Float16*)alloc((size_t)cM * cHDK * 2);
    _Float16* VT  = (_Float16*)alloc((size_t)cM * cHDK * 2);
    _Float16* OH  = (_Float16*)alloc((size_t)cM * cHDK * 2);
    _Float16* PH  = (_Float16*)alloc((size_t)cB * cH * cS * cS * 2);
    _Float16* FH  = (_Float16*)alloc((size_t)cM * cF * 2);
    _Float16* wq1t = (_Float16*)alloc((size_t)cL * cHDK * cD * 2);
    _Float16* wk1t = (_Float16*)alloc((size_t)cL * cHDK * cD * 2);
    _Float16* wv1t = (_Float16*)alloc((size_t)cL * cHDK * cD * 2);
    _Float16* wo1t = (_Float16*)alloc((size_t)cL * cD * cHDK * 2);
    _Float16* wq2t = (_Float16*)alloc((size_t)cL * cHDK * cD * 2);
    _Float16* wk2t = (_Float16*)alloc((size_t)cL * cHDK * cD * 2);
    _Float16* wv2t = (_Float16*)alloc((size_t)cL * cHDK * cD * 2);
    _Float16* wo2t = (_Float16*)alloc((size_t)cL * cD * cHDK * 2);
    _Float16* w1ft = (_Float16*)alloc((size_t)cL * cF * cD * 2);
    _Float16* w2ft = (_Float16*)alloc((size_t)cL * cD * cF * 2);

    // ---- GEMM launcher (block tile: 128 rows x 64 cols) ----
    auto gemm = [&](const _Float16* A, const _Float16* Bt, float* C, _Float16* Ch,
                    const float* bias, int Mm, int Nn, int Kk,
                    int lda, int ldb, int ldc,
                    long sAo, long sAi, long sBo, long sBi, long sCo, long sCi,
                    int innerB, int batches, float alpha, int act) {
        dim3 grid(Nn / 64, Mm / 128, batches);
        k_gemm_wmma<<<grid, 128, 0, stream>>>(
            A, Bt, C, Ch, bias, Kk, lda, ldb, ldc,
            sAo, sAi, sBo, sBi, sCo, sCi, innerB, alpha, act);
    };

    // ---- one-time weight / input prep ----
    {
        int t1 = cL * cH * cD * cDK;
        int blk = 256;
        k_prep_hdk<<<(t1 + blk - 1) / blk, blk, 0, stream>>>(wq1, wq1t, t1);
        k_prep_hdk<<<(t1 + blk - 1) / blk, blk, 0, stream>>>(wk1, wk1t, t1);
        k_prep_hdk<<<(t1 + blk - 1) / blk, blk, 0, stream>>>(wv1, wv1t, t1);
        k_prep_hdk<<<(t1 + blk - 1) / blk, blk, 0, stream>>>(wq2, wq2t, t1);
        k_prep_hdk<<<(t1 + blk - 1) / blk, blk, 0, stream>>>(wk2, wk2t, t1);
        k_prep_hdk<<<(t1 + blk - 1) / blk, blk, 0, stream>>>(wv2, wv2t, t1);
        int t2 = cL * cHDK * cD;
        k_prep_kn<<<(t2 + blk - 1) / blk, blk, 0, stream>>>(wo1, wo1t, cHDK, cD, t2);
        k_prep_kn<<<(t2 + blk - 1) / blk, blk, 0, stream>>>(wo2, wo2t, cHDK, cD, t2);
        int t3 = cL * cD * cF;
        k_prep_kn<<<(t3 + blk - 1) / blk, blk, 0, stream>>>(w1f, w1ft, cD, cF, t3);
        k_prep_kn<<<(t3 + blk - 1) / blk, blk, 0, stream>>>(w2f, w2ft, cF, cD, t3);
        int t4 = cM * cD;
        k_cvt<<<(t4 + blk - 1) / blk, blk, 0, stream>>>(memory, MH, t4);
        k_embed<<<(t4 + blk - 1) / blk, blk, 0, stream>>>(
            tgt, maskp, w_f2h, b_f2h, timei, gender, race,
            emb_time, emb_gen, emb_race, X, XH);
    }

    const int vtBlocks = (cM * cHDK + 255) / 256;

    // attention sub-pipeline (q/k sources already projected to QH/KH, V in VH)
    auto attention = [&](const _Float16* Wo, const float* boL) {
        k_vt<<<vtBlocks, 256, 0, stream>>>(VH, VT);
        // scores[b,h] = Q[b,h] @ K[b,h]^T / 8
        gemm(QH, KH, SC, nullptr, nullptr, cS, cS, cDK,
             cHDK, cHDK, cS,
             (long)cS * cHDK, cDK, (long)cS * cHDK, cDK,
             (long)cH * cS * cS, (long)cS * cS,
             cH, cB * cH, 0.125f, 0);
        k_softmax<<<cB * cH * cS, 32, 0, stream>>>(SC, PH);
        // O[b,h] = P[b,h] @ V[b,h]   (heads concatenated via strides)
        gemm(PH, VT, nullptr, OH, nullptr, cS, cDK, cS,
             cS, cS, cHDK,
             (long)cH * cS * cS, (long)cS * cS,
             (long)cH * cDK * cS, (long)cDK * cS,
             (long)cS * cHDK, cDK,
             cH, cB * cH, 1.f, 0);
        // output projection -> Y
        gemm(OH, Wo, Y, nullptr, boL, cM, cD, cHDK,
             cHDK, cHDK, cD, 0, 0, 0, 0, 0, 0, 1, 1, 1.f, 0);
    };

    for (int l = 0; l < cL; ++l) {
        const size_t wQKV = (size_t)l * cHDK * cD;
        const size_t wOUT = (size_t)l * cD * cHDK;
        const size_t wF1  = (size_t)l * cF * cD;
        const size_t wF2  = (size_t)l * cD * cF;

        // ---- self-attention block: x = LN(x + MHA(x,x,x)) ----
        gemm(XH, wq1t + wQKV, nullptr, QH, bq1 + l * cHDK, cM, cHDK, cD,
             cD, cD, cHDK, 0, 0, 0, 0, 0, 0, 1, 1, 1.f, 0);
        gemm(XH, wk1t + wQKV, nullptr, KH, bk1 + l * cHDK, cM, cHDK, cD,
             cD, cD, cHDK, 0, 0, 0, 0, 0, 0, 1, 1, 1.f, 0);
        gemm(XH, wv1t + wQKV, nullptr, VH, bv1 + l * cHDK, cM, cHDK, cD,
             cD, cD, cHDK, 0, 0, 0, 0, 0, 0, 1, 1, 1.f, 0);
        attention(wo1t + wOUT, bo1 + l * cD);
        k_add_ln<<<cM, 32, 0, stream>>>(X, Y, g1 + l * cD, be1 + l * cD, XH);

        // ---- cross block: x = LN(x + MHA(memory, memory, x)) ----
        gemm(MH, wq2t + wQKV, nullptr, QH, bq2 + l * cHDK, cM, cHDK, cD,
             cD, cD, cHDK, 0, 0, 0, 0, 0, 0, 1, 1, 1.f, 0);
        gemm(MH, wk2t + wQKV, nullptr, KH, bk2 + l * cHDK, cM, cHDK, cD,
             cD, cD, cHDK, 0, 0, 0, 0, 0, 0, 1, 1, 1.f, 0);
        gemm(XH, wv2t + wQKV, nullptr, VH, bv2 + l * cHDK, cM, cHDK, cD,
             cD, cD, cHDK, 0, 0, 0, 0, 0, 0, 1, 1, 1.f, 0);
        attention(wo2t + wOUT, bo2 + l * cD);
        k_add_ln<<<cM, 32, 0, stream>>>(X, Y, g2 + l * cD, be2 + l * cD, XH);

        // ---- FFN block: x = LN(x + relu(x@W1+b1)@W2+b2) ----
        gemm(XH, w1ft + wF1, nullptr, FH, b1f + l * cF, cM, cF, cD,
             cD, cD, cF, 0, 0, 0, 0, 0, 0, 1, 1, 1.f, 1 /*relu*/);
        gemm(FH, w2ft + wF2, Y, nullptr, b2f + l * cD, cM, cD, cF,
             cF, cF, cD, 0, 0, 0, 0, 0, 0, 1, 1, 1.f, 0);
        k_add_ln<<<cM, 32, 0, stream>>>(X, Y, gf + l * cD, bff + l * cD, XH);
    }

    // ---- heads + pass-through of `time` ----
    float* out_p  = (float*)d_out;                       // [B,S,FEAT]
    float* time_p = out_p + (size_t)cM * cFEAT;          // [B,S] (int bits)
    float* mout_p = time_p + cM;                         // [B,S,FEAT]

    hipMemcpyAsync(time_p, timei, (size_t)cM * sizeof(int),
                   hipMemcpyDeviceToDevice, stream);

    int hTot = cM * cFEAT * 2;
    k_head<<<(hTot + 255) / 256, 256, 0, stream>>>(
        X, w_out, b_out, w_m, b_m, out_p, mout_p);
}
